// MultiHeadAttention_88029649699660
// MI455X (gfx1250) — matrix operand order repaired and checked by reading, not compiled, not run on hardware
//
#include <hip/hip_runtime.h>
#include <hip/hip_bf16.h>

// ---------------------------------------------------------------------------
// Multi-head attention (B=2, S=2048, D=1024, H=16, Dh=64) for gfx1250.
// f16 WMMA (v_wmma_f32_16x16x32_f16), f32 accumulation, f32 softmax.
// - scores->softmax->attn@V fused per (b,h,q-tile); 16x2048 score strip lives
//   in LDS, attn crosses HBM exactly once (roofline minimum).
// - projection GEMMs register-blocked 2x4 (32x64 per wave) to cut L2 traffic
//   2.7x and deepen the per-wave WMMA chain.
// ---------------------------------------------------------------------------

typedef _Float16 f16;
typedef _Float16 v16h  __attribute__((ext_vector_type(16)));
typedef _Float16 v8h   __attribute__((ext_vector_type(8)));
typedef float    v8f   __attribute__((ext_vector_type(8)));
typedef float    f32x4 __attribute__((ext_vector_type(4)));

#define BATCH   2
#define SEQ     2048
#define DMODEL  1024
#define NHEADS  16
#define HDIM    64
#define SCALE_F 0.125f   // 1/sqrt(64)

// LDS layout for the fused attention kernel (dynamic shared memory)
#define PSTRIDE 2052                         // 2048 + 4 pad -> conflict-free frags
#define P_BYTES   (16 * PSTRIDE * 4)         // 131328: exp(score) strip, f32
#define CP_BYTES  (8 * 16 * HDIM * 4)        // 32768 : per-wave ctx partials
#define RED_BYTES (256 * 4)                  // 1024  : reduction scratch
#define RM_BYTES  (16 * 4)                   // rowmax
#define RI_BYTES  (16 * 4)                   // rowinv
#define SMEM_BYTES (P_BYTES + CP_BYTES + RED_BYTES + RM_BYTES + RI_BYTES)

// ---------------------------------------------------------------------------
// WMMA fragment builders (wave32 layouts per CDNA5 ISA 7.12.2)
// A (16x32 f16): lane&15 = M row; lane<16 -> K {0..7,16..23}, lane>=16 -> {8..15,24..31}
// B (32x16 f16): lane&15 = N col; lane<16 -> K {0..7,16..23}, lane>=16 -> {8..15,24..31}
// C/D (16x16 f32): lane&15 = N col; VGPR r -> M = 8*(lane>=16) + r
// ---------------------------------------------------------------------------

static __device__ __forceinline__ v16h frag_a_f32(const float* row, int k0) {
  const unsigned lane = threadIdx.x & 31u;
  const int klo = (int)((lane >> 4) << 3);                 // 0 or 8
  const f32x4* p0 = (const f32x4*)(row + k0 + klo);        // K = klo+0..7
  const f32x4* p1 = (const f32x4*)(row + k0 + klo + 16);   // K = klo+16..23
  f32x4 x0 = p0[0], x1 = p0[1], x2 = p1[0], x3 = p1[1];
  v16h a;
#pragma unroll
  for (int j = 0; j < 4; ++j) {
    a[j]      = (f16)x0[j];
    a[4 + j]  = (f16)x1[j];
    a[8 + j]  = (f16)x2[j];
    a[12 + j] = (f16)x3[j];
  }
  return a;
}

static __device__ __forceinline__ v16h frag_a_f16(const f16* __restrict__ row, int k0) {
  const unsigned lane = threadIdx.x & 31u;
  const int klo = (int)((lane >> 4) << 3);
  v8h lo = *(const v8h*)(row + k0 + klo);
  v8h hi = *(const v8h*)(row + k0 + klo + 16);
  v16h a;
#pragma unroll
  for (int j = 0; j < 8; ++j) { a[j] = lo[j]; a[8 + j] = hi[j]; }
  return a;
}

static __device__ __forceinline__ v16h frag_b_f32(const float* __restrict__ row, int k0) {
  const unsigned lane = threadIdx.x & 31u;
  const int kb = (int)((lane >> 4) << 3);                  // 0 or 8
  const f32x4* p0 = (const f32x4*)(row + k0 + kb);
  const f32x4* p1 = (const f32x4*)(row + k0 + kb + 16);
  f32x4 x0 = p0[0], x1 = p0[1], x2 = p1[0], x3 = p1[1];
  v16h b;
#pragma unroll
  for (int j = 0; j < 4; ++j) {
    b[j]      = (f16)x0[j];
    b[4 + j]  = (f16)x1[j];
    b[8 + j]  = (f16)x2[j];
    b[12 + j] = (f16)x3[j];
  }
  return b;
}

static __device__ __forceinline__ v16h frag_b_f16(const f16* __restrict__ row, int k0) {
  const unsigned lane = threadIdx.x & 31u;
  const int kb = (int)((lane >> 4) << 3);
  v8h lo = *(const v8h*)(row + k0 + kb);
  v8h hi = *(const v8h*)(row + k0 + kb + 16);
  v16h b;
#pragma unroll
  for (int j = 0; j < 8; ++j) { b[j] = lo[j]; b[8 + j] = hi[j]; }
  return b;
}

static __device__ __forceinline__ v8f wmma_f16(v16h a, v16h b, v8f c) {
  return __builtin_amdgcn_wmma_f32_16x16x32_f16(false, a, false, b, (short)0, c,
                                                false, false);
}

// ---------------------------------------------------------------------------
// 1) Fused Q/K/V projection: Y = X @ W^T + bias, head-major f16.
//    Register-blocked: each wave owns a 32x64 output block (2x4 tiles,
//    8 accumulators, 8 WMMAs per 32-deep k-step).
//    V written pre-transposed: Vt[b,h,dh,s].
//    Jobs: 3 * B * (S/32) * (D/64) = 6144 waves -> 768 blocks.
// ---------------------------------------------------------------------------
__global__ __launch_bounds__(256)
void proj_qkv_kernel(const float* __restrict__ Xq, const float* __restrict__ Xk,
                     const float* __restrict__ Xv,
                     const float* __restrict__ Wq, const float* __restrict__ Wk,
                     const float* __restrict__ Wv,
                     const float* __restrict__ bq, const float* __restrict__ bk,
                     const float* __restrict__ bv,
                     f16* __restrict__ Qh, f16* __restrict__ Kh,
                     f16* __restrict__ Vt) {
  unsigned wid = blockIdx.x * 8u + (threadIdx.x >> 5);
  const unsigned nt = wid & 15u; wid >>= 4;       // 16 x 64-wide col blocks
  const unsigned mt = wid & 63u; wid >>= 6;       // 64 x 32-high row blocks
  const unsigned b  = wid & 1u;
  const unsigned which = wid >> 1;                // 0=Q 1=K 2=V (wave-uniform)

  const float* X    = (which == 0) ? Xq : (which == 1) ? Xk : Xv;
  const float* W    = (which == 0) ? Wq : (which == 1) ? Wk : Wv;
  const float* bias = (which == 0) ? bq : (which == 1) ? bk : bv;

  const unsigned lane = threadIdx.x & 31u;
  const int l15 = (int)(lane & 15u);
  const int m0 = (int)mt * 32;
  const int o0 = (int)nt * 64;

  const float* arow0 = X + ((size_t)b * SEQ + (size_t)(m0 + l15)) * DMODEL;
  const float* arow1 = arow0 + (size_t)16 * DMODEL;
  const float* brow0 = W + (size_t)(o0 + l15) * DMODEL;
  const float* brow1 = brow0 + (size_t)16 * DMODEL;
  const float* brow2 = brow0 + (size_t)32 * DMODEL;
  const float* brow3 = brow0 + (size_t)48 * DMODEL;

  v8f acc[2][4];
#pragma unroll
  for (int mi = 0; mi < 2; ++mi)
#pragma unroll
    for (int nj = 0; nj < 4; ++nj) acc[mi][nj] = (v8f){};

  for (int k0 = 0; k0 < DMODEL; k0 += 32) {
    const v16h a0 = frag_a_f32(arow0, k0);
    const v16h a1 = frag_a_f32(arow1, k0);
    v16h bf;
    bf = frag_b_f32(brow0, k0);
    acc[0][0] = wmma_f16(a0, bf, acc[0][0]);
    acc[1][0] = wmma_f16(a1, bf, acc[1][0]);
    bf = frag_b_f32(brow1, k0);
    acc[0][1] = wmma_f16(a0, bf, acc[0][1]);
    acc[1][1] = wmma_f16(a1, bf, acc[1][1]);
    bf = frag_b_f32(brow2, k0);
    acc[0][2] = wmma_f16(a0, bf, acc[0][2]);
    acc[1][2] = wmma_f16(a1, bf, acc[1][2]);
    bf = frag_b_f32(brow3, k0);
    acc[0][3] = wmma_f16(a0, bf, acc[0][3]);
    acc[1][3] = wmma_f16(a1, bf, acc[1][3]);
  }

  const int n  = l15;
  const int mb = (int)((lane >> 4) << 3);
#pragma unroll
  for (int nj = 0; nj < 4; ++nj) {
    const int o  = o0 + nj * 16 + n;
    const int h  = o >> 6;
    const int dh = o & 63;
    const float bb = bias[o];
    if (which == 2) {                              // V, transposed layout
      f16* dst = Vt + (((size_t)b * NHEADS + h) * HDIM + dh) * SEQ;
#pragma unroll
      for (int mi = 0; mi < 2; ++mi)
#pragma unroll
        for (int r = 0; r < 8; ++r)
          dst[m0 + mi * 16 + mb + r] = (f16)(acc[mi][nj][r] + bb);
    } else {                                       // Q or K, [B,H,S,Dh]
      f16* dst = ((which == 0) ? Qh : Kh) +
                 (((size_t)b * NHEADS + h) * SEQ) * HDIM + dh;
#pragma unroll
      for (int mi = 0; mi < 2; ++mi)
#pragma unroll
        for (int r = 0; r < 8; ++r)
          dst[(size_t)(m0 + mi * 16 + mb + r) * HDIM] = (f16)(acc[mi][nj][r] + bb);
    }
  }
}

// ---------------------------------------------------------------------------
// 2) RoPE on Q and K in-place (math in f32).
// ---------------------------------------------------------------------------
__global__ __launch_bounds__(256)
void rope_kernel(f16* __restrict__ Qh, f16* __restrict__ Kh,
                 const float* __restrict__ cosb, const float* __restrict__ sinb) {
  const size_t TOT = (size_t)BATCH * NHEADS * SEQ * (HDIM / 2);
  size_t id = (size_t)blockIdx.x * blockDim.x + threadIdx.x;   // 0 .. 2*TOT-1
  f16* base;
  if (id >= TOT) { id -= TOT; base = Kh; } else { base = Qh; }
  const int    p  = (int)(id & 31u);
  const int    s  = (int)((id >> 5) & 2047u);
  const size_t bh = id >> 16;
  f16* x = base + ((size_t)bh * SEQ + s) * HDIM + 2 * p;
  const float xe = (float)x[0];
  const float xo = (float)x[1];
  const float ce = cosb[(size_t)s * HDIM + 2 * p];
  const float co = cosb[(size_t)s * HDIM + 2 * p + 1];
  const float se = sinb[(size_t)s * HDIM + 2 * p];
  const float so = sinb[(size_t)s * HDIM + 2 * p + 1];
  x[0] = (f16)(xe * ce - xo * so);
  x[1] = (f16)(xo * co + xe * se);
}

// ---------------------------------------------------------------------------
// 3) FUSED attention: one block per (b*h, 16-row q-tile). 4096 blocks.
//    Phase 1: scores strip (16 x 2048) -> LDS (f32, WMMA)
//    Phase 2: rowmax / exp / rowsum in LDS; write normalized attn f32 to HBM
//             exactly once; keep unnormalized exp in LDS, fold 1/sum later.
//    Phase 3: ctx = P @ V. Each wave: 256-deep K slice x all 4 d-tiles
//             (A-frag from LDS reused 4x); partials reduced through LDS.
// ---------------------------------------------------------------------------
__global__ __launch_bounds__(256)
void fused_attn_kernel(const f16* __restrict__ Qh, const f16* __restrict__ Kh,
                       const f16* __restrict__ Vt,
                       float* __restrict__ attn, f16* __restrict__ ctx) {
  extern __shared__ __align__(16) char smem_raw[];
  float* P      = (float*)smem_raw;                         // [16][PSTRIDE]
  float* CP     = (float*)(smem_raw + P_BYTES);             // [8][16][64]
  float* red    = (float*)(smem_raw + P_BYTES + CP_BYTES);  // [256]
  float* rowmax = (float*)(smem_raw + P_BYTES + CP_BYTES + RED_BYTES);
  float* rowinv = (float*)(smem_raw + P_BYTES + CP_BYTES + RED_BYTES + RM_BYTES);

  const unsigned t    = threadIdx.x;
  const unsigned wave = t >> 5;
  const unsigned lane = t & 31u;
  const unsigned bh   = blockIdx.x >> 7;          // 0..31
  const unsigned qt   = blockIdx.x & 127u;
  const int q0 = (int)qt * 16;

  // ---- Phase 1: scores into LDS --------------------------------------------
  const f16* qrow = Qh + ((size_t)bh * SEQ + (size_t)(q0 + (int)(lane & 15u))) * HDIM;
  const v16h aQ0 = frag_a_f16(qrow, 0);
  const v16h aQ1 = frag_a_f16(qrow, 32);
  const int n  = (int)(lane & 15u);
  const int mb = (int)((lane >> 4) << 3);

  for (int i = 0; i < 16; ++i) {
    const int kt = (int)wave * 16 + i;            // k-tile 0..127
    const int c0 = kt * 16;
    const f16* krow = Kh + ((size_t)bh * SEQ + (size_t)(c0 + n)) * HDIM;
    v8f c = {};
    c = wmma_f16(aQ0, frag_b_f16(krow, 0),  c);
    c = wmma_f16(aQ1, frag_b_f16(krow, 32), c);
#pragma unroll
    for (int r = 0; r < 8; ++r)
      P[(mb + r) * PSTRIDE + c0 + n] = c[r] * SCALE_F;
  }
  __syncthreads();

  // ---- Phase 2a: row max ----------------------------------------------------
  const int row   = (int)(t >> 4);                // 0..15
  const int chunk = (int)(t & 15u);               // 0..15
  {
    float m = -3.4e38f;
    for (int j = 0; j < 128; ++j)
      m = fmaxf(m, P[row * PSTRIDE + chunk + 16 * j]);
    red[t] = m;
  }
  __syncthreads();
  if (t < 16) {
    float m = -3.4e38f;
#pragma unroll
    for (int j = 0; j < 16; ++j) m = fmaxf(m, red[t * 16 + j]);
    rowmax[t] = m;
  }
  __syncthreads();

  // ---- Phase 2b: exp + row sum (exp left in LDS, unnormalized) -------------
  {
    const float m = rowmax[row];
    float s = 0.f;
    for (int j = 0; j < 128; ++j) {
      const int idx = row * PSTRIDE + chunk + 16 * j;
      const float e = __expf(P[idx] - m);
      P[idx] = e;
      s += e;
    }
    red[t] = s;
  }
  __syncthreads();
  if (t < 16) {
    float s = 0.f;
#pragma unroll
    for (int j = 0; j < 16; ++j) s += red[t * 16 + j];
    rowinv[t] = 1.0f / s;
  }
  __syncthreads();

  // ---- Phase 2c: coalesced single write of normalized attn -----------------
  {
    float* abase = attn + ((size_t)bh * SEQ + q0) * SEQ;
    for (int idx = (int)t; idx < 16 * SEQ; idx += 256) {
      const int rr = idx >> 11;                   // 0..15
      const int cc = idx & 2047;
      abase[(size_t)rr * SEQ + cc] = P[rr * PSTRIDE + cc] * rowinv[rr];
    }
  }
  __syncthreads();

  // ---- Phase 3: ctx = P @ V (wave = 256-deep K slice, all 4 d-tiles) -------
  v8f acc0 = {}, acc1 = {}, acc2 = {}, acc3 = {};
  {
    const float* prow = P + (lane & 15u) * PSTRIDE;
    const f16* vrow0 = Vt + ((size_t)bh * HDIM +  0 + n) * SEQ;
    const f16* vrow1 = Vt + ((size_t)bh * HDIM + 16 + n) * SEQ;
    const f16* vrow2 = Vt + ((size_t)bh * HDIM + 32 + n) * SEQ;
    const f16* vrow3 = Vt + ((size_t)bh * HDIM + 48 + n) * SEQ;
    const int kbeg = (int)wave * 256;
    for (int ks = 0; ks < 8; ++ks) {
      const int k0 = kbeg + ks * 32;
      const v16h a = frag_a_f32(prow, k0);        // ds_load_b128, conflict-free
      acc0 = wmma_f16(a, frag_b_f16(vrow0, k0), acc0);
      acc1 = wmma_f16(a, frag_b_f16(vrow1, k0), acc1);
      acc2 = wmma_f16(a, frag_b_f16(vrow2, k0), acc2);
      acc3 = wmma_f16(a, frag_b_f16(vrow3, k0), acc3);
    }
  }
  {
    float* CPw = CP + wave * (16 * HDIM);
#pragma unroll
    for (int r = 0; r < 8; ++r) {
      CPw[(mb + r) * HDIM +  0 + n] = acc0[r];
      CPw[(mb + r) * HDIM + 16 + n] = acc1[r];
      CPw[(mb + r) * HDIM + 32 + n] = acc2[r];
      CPw[(mb + r) * HDIM + 48 + n] = acc3[r];
    }
  }
  __syncthreads();

  // ---- Final: reduce 8 wave partials, scale by 1/rowsum, store ctx f16 -----
  {
    const unsigned b = bh >> 4, h = bh & 15u;
    const int idx = (int)t * 4;                   // 1024 elements, 4 per thread
    const int m   = idx >> 6;                     // row in tile
    const int cc  = idx & 63;                     // dh base (aligned 4)
    f32x4 s = {};
#pragma unroll
    for (int w = 0; w < 8; ++w)
      s += *(const f32x4*)(CP + w * (16 * HDIM) + m * HDIM + cc);
    const float inv = rowinv[m];
    f16* dst = ctx + ((size_t)b * SEQ + (size_t)(q0 + m)) * DMODEL + h * HDIM + cc;
#pragma unroll
    for (int j = 0; j < 4; ++j) dst[j] = (f16)(s[j] * inv);
  }
}

// ---------------------------------------------------------------------------
// 4) out = ctx @ Wo^T + bo (f32). Register-blocked 2x4 like proj:
//    jobs = B * (S/32) * (D/64) = 2048 waves -> 256 blocks.
// ---------------------------------------------------------------------------
__global__ __launch_bounds__(256)
void out_proj_kernel(const f16* __restrict__ ctx, const float* __restrict__ Wo,
                     const float* __restrict__ bo, float* __restrict__ out) {
  unsigned wid = blockIdx.x * 8u + (threadIdx.x >> 5);
  const unsigned nt = wid & 15u; wid >>= 4;
  const unsigned mt = wid & 63u; wid >>= 6;
  const unsigned b  = wid;                         // 0..1
  const unsigned lane = threadIdx.x & 31u;
  const int l15 = (int)(lane & 15u);
  const int m0 = (int)mt * 32;
  const int o0 = (int)nt * 64;

  const f16* arow0 = ctx + ((size_t)b * SEQ + (size_t)(m0 + l15)) * DMODEL;
  const f16* arow1 = arow0 + (size_t)16 * DMODEL;
  const float* brow0 = Wo + (size_t)(o0 + l15) * DMODEL;
  const float* brow1 = brow0 + (size_t)16 * DMODEL;
  const float* brow2 = brow0 + (size_t)32 * DMODEL;
  const float* brow3 = brow0 + (size_t)48 * DMODEL;

  v8f acc[2][4];
#pragma unroll
  for (int mi = 0; mi < 2; ++mi)
#pragma unroll
    for (int nj = 0; nj < 4; ++nj) acc[mi][nj] = (v8f){};

  for (int k0 = 0; k0 < DMODEL; k0 += 32) {
    const v16h a0 = frag_a_f16(arow0, k0);
    const v16h a1 = frag_a_f16(arow1, k0);
    v16h bf;
    bf = frag_b_f32(brow0, k0);
    acc[0][0] = wmma_f16(a0, bf, acc[0][0]);
    acc[1][0] = wmma_f16(a1, bf, acc[1][0]);
    bf = frag_b_f32(brow1, k0);
    acc[0][1] = wmma_f16(a0, bf, acc[0][1]);
    acc[1][1] = wmma_f16(a1, bf, acc[1][1]);
    bf = frag_b_f32(brow2, k0);
    acc[0][2] = wmma_f16(a0, bf, acc[0][2]);
    acc[1][2] = wmma_f16(a1, bf, acc[1][2]);
    bf = frag_b_f32(brow3, k0);
    acc[0][3] = wmma_f16(a0, bf, acc[0][3]);
    acc[1][3] = wmma_f16(a1, bf, acc[1][3]);
  }

  const int n  = l15;
  const int mb = (int)((lane >> 4) << 3);
#pragma unroll
  for (int nj = 0; nj < 4; ++nj) {
    const int o = o0 + nj * 16 + n;
    const float bb = bo[o];
#pragma unroll
    for (int mi = 0; mi < 2; ++mi) {
      float* dst = out + ((size_t)b * SEQ + (size_t)(m0 + mi * 16 + mb)) * DMODEL + o;
#pragma unroll
      for (int r = 0; r < 8; ++r) dst[(size_t)r * DMODEL] = acc[mi][nj][r] + bb;
    }
  }
}

// ---------------------------------------------------------------------------
// Host launcher
// ---------------------------------------------------------------------------
extern "C" void kernel_launch(void* const* d_in, const int* in_sizes, int n_in,
                              void* d_out, int out_size, void* d_ws, size_t ws_size,
                              hipStream_t stream) {
  (void)in_sizes; (void)n_in; (void)out_size; (void)ws_size;

  const float* query = (const float*)d_in[0];
  const float* key   = (const float*)d_in[1];
  const float* value = (const float*)d_in[2];
  const float* cosb  = (const float*)d_in[3];
  const float* sinb  = (const float*)d_in[4];
  const float* Wq    = (const float*)d_in[5];
  const float* bq    = (const float*)d_in[6];
  const float* Wk    = (const float*)d_in[7];
  const float* bk    = (const float*)d_in[8];
  const float* Wv    = (const float*)d_in[9];
  const float* bv    = (const float*)d_in[10];
  const float* Wo    = (const float*)d_in[11];
  const float* bo    = (const float*)d_in[12];

  float* out  = (float*)d_out;                                    // [B,S,D]
  float* attn = out + (size_t)BATCH * SEQ * DMODEL;               // [B,H,S,S]

  const size_t QKV = (size_t)BATCH * NHEADS * SEQ * HDIM;         // 4,194,304 halves
  f16* Qh  = (f16*)d_ws;                                          //  8 MB
  f16* Kh  = Qh + QKV;                                            //  8 MB
  f16* Vt  = Kh + QKV;                                            //  8 MB  [B,H,Dh,S]
  f16* ctx = Vt + QKV;                                            //  8 MB  [B,S,D]

  dim3 blk(256);

  // 1) Q/K/V projections: 6144 register-blocked wave-jobs / 8 per block
  proj_qkv_kernel<<<dim3(768), blk, 0, stream>>>(query, key, value,
                                                 Wq, Wk, Wv, bq, bk, bv,
                                                 Qh, Kh, Vt);
  // 2) RoPE on Q and K
  rope_kernel<<<dim3(16384), blk, 0, stream>>>(Qh, Kh, cosb, sinb);
  // 3) fused scores+softmax+attn@V: one block per (b*h, q-tile)
  fused_attn_kernel<<<dim3(4096), blk, SMEM_BYTES, stream>>>(Qh, Kh, Vt,
                                                             attn, ctx);
  // 4) output projection: 2048 register-blocked wave-jobs / 8 per block
  out_proj_kernel<<<dim3(256), blk, 0, stream>>>(ctx, Wo, bo, out);
}
